// SelfAttention_64390149702099
// MI455X (gfx1250) — matrix operand order repaired and checked by reading, not compiled, not run on hardware
//
#include <hip/hip_runtime.h>

typedef __attribute__((ext_vector_type(16))) __bf16 v16bf;
typedef __attribute__((ext_vector_type(8)))  __bf16 v8bf;
typedef __attribute__((ext_vector_type(4)))  __bf16 v4bf;
typedef __attribute__((ext_vector_type(8)))  float  v8f;

static constexpr int Bb = 8, Ss = 2048, Dd = 512, Ee = 512;
static constexpr int MM = Bb * Ss; // 16384

// Workspace layout (bytes)
static constexpr size_t OFF_XBF = 0;                          // X as bf16
static constexpr size_t SZ_XBF  = (size_t)MM * Dd * 2;        // 16.78 MB
static constexpr size_t OFF_WT  = OFF_XBF + SZ_XBF;           // W^T bf16 x3
static constexpr size_t SZ_WT1  = (size_t)Ee * Dd * 2;        // 512 KB each
static constexpr size_t OFF_Q   = OFF_WT + 3 * SZ_WT1;
static constexpr size_t SZ_QKV  = (size_t)MM * Ee * 2;        // 16.78 MB
static constexpr size_t OFF_K   = OFF_Q + SZ_QKV;
static constexpr size_t OFF_VT  = OFF_K + SZ_QKV;             // V^T [b][e][s]

#define LOG2E 1.44269504088896340736f

__device__ __forceinline__ v16bf cat16(v8bf lo, v8bf hi) {
    return __builtin_shufflevector(lo, hi, 0,1,2,3,4,5,6,7,8,9,10,11,12,13,14,15);
}
__device__ __forceinline__ float rmax16(float v) {
    v = fmaxf(v, __shfl_xor(v, 1));
    v = fmaxf(v, __shfl_xor(v, 2));
    v = fmaxf(v, __shfl_xor(v, 4));
    v = fmaxf(v, __shfl_xor(v, 8));
    return v;
}
__device__ __forceinline__ float rsum16(float v) {
    v += __shfl_xor(v, 1);
    v += __shfl_xor(v, 2);
    v += __shfl_xor(v, 4);
    v += __shfl_xor(v, 8);
    return v;
}

// CDNA5 async copy: 16B global -> LDS, tracked by ASYNCcnt (no VGPR round-trip).
__device__ __forceinline__ void async_copy16(uint32_t lds_addr, const __bf16* gsrc) {
    asm volatile("global_load_async_to_lds_b128 %0, %1, off"
                 :: "v"(lds_addr), "v"((unsigned long long)(uintptr_t)gsrc)
                 : "memory");
}
__device__ __forceinline__ void wait_async_all() {
    asm volatile("s_wait_asynccnt 0x0" ::: "memory");
}

// ---------- Kernel 1: X fp32 -> bf16 ----------
__global__ __launch_bounds__(256) void cvt_x_kernel(const float* __restrict__ x,
                                                    __bf16* __restrict__ xb, int n4) {
    int i = blockIdx.x * 256 + threadIdx.x;
    if (i >= n4) return;
    float4 f = ((const float4*)x)[i];
    v4bf o;
    o[0] = (__bf16)f.x; o[1] = (__bf16)f.y; o[2] = (__bf16)f.z; o[3] = (__bf16)f.w;
    ((v4bf*)xb)[i] = o;
}

// ---------- Kernel 2: W [D][E] fp32 -> W^T [E][D] bf16 ----------
__global__ __launch_bounds__(256) void cvt_wt_kernel(const float* __restrict__ wq,
                                                     const float* __restrict__ wk,
                                                     const float* __restrict__ wv,
                                                     __bf16* __restrict__ wt) {
    int n = blockIdx.x;            // output column of W == row of W^T
    int z = blockIdx.y;            // which matrix
    const float* w = (z == 0) ? wq : (z == 1) ? wk : wv;
    __bf16* o = wt + (size_t)z * Ee * Dd + (size_t)n * Dd;
    for (int k = threadIdx.x; k < Dd; k += 256)
        o[k] = (__bf16)w[(size_t)k * Ee + n];
}

// ---------- Kernel 3: QKV projection GEMM (bf16 WMMA, fp32 acc) ----------
// C[m][n] = sum_k Xbf[m][k] * WT[n][k]; tile 128x128, 8 waves, K-step 32,
// double-buffered async-to-LDS staging.
__global__ __launch_bounds__(256) void qkv_gemm_kernel(const __bf16* __restrict__ xb,
                                                       const __bf16* __restrict__ wt,
                                                       __bf16* __restrict__ qout,
                                                       __bf16* __restrict__ kout,
                                                       __bf16* __restrict__ vtout) {
    constexpr int LDA = 40; // padded stride (elements) to break bank conflicts
    __shared__ __bf16 As[2][128 * LDA];
    __shared__ __bf16 Bs[2][128 * LDA];

    const int z  = blockIdx.z;
    const int n0 = blockIdx.x * 128;
    const int m0 = blockIdx.y * 128;
    const __bf16* wz = wt + (size_t)z * Ee * Dd;

    const int tid  = threadIdx.x;
    const int lane = tid & 31;
    const int w    = tid >> 5;      // 8 waves
    const int wm   = w & 3;         // 4 waves along M (32 rows each)
    const int wn   = w >> 2;        // 2 waves along N (64 cols each)
    const int lr   = lane & 15;
    const int lh   = lane >> 4;
    const int kbse = lh * 8;        // A-frag K base
    const int klo  = lh * 16;       // unused; B-frag also uses kbse

    // this thread's two 16B staging chunks (row/sub fixed across steps)
    const int c0 = tid, c1 = tid + 256;
    const int row0 = c0 >> 2, sub0 = c0 & 3;
    const int row1 = c1 >> 2, sub1 = c1 & 3;

    v8f acc[2][4] = {};

    auto stage = [&](int p, int k0) {
        async_copy16((uint32_t)(uintptr_t)&As[p][row0 * LDA + sub0 * 8],
                     &xb[(size_t)(m0 + row0) * Dd + k0 + sub0 * 8]);
        async_copy16((uint32_t)(uintptr_t)&Bs[p][row0 * LDA + sub0 * 8],
                     &wz[(size_t)(n0 + row0) * Dd + k0 + sub0 * 8]);
        async_copy16((uint32_t)(uintptr_t)&As[p][row1 * LDA + sub1 * 8],
                     &xb[(size_t)(m0 + row1) * Dd + k0 + sub1 * 8]);
        async_copy16((uint32_t)(uintptr_t)&Bs[p][row1 * LDA + sub1 * 8],
                     &wz[(size_t)(n0 + row1) * Dd + k0 + sub1 * 8]);
    };

    stage(0, 0);
    wait_async_all();
    __syncthreads();

    constexpr int NSTEP = Dd / 32; // 16
    for (int step = 0; step < NSTEP; ++step) {
        const int p = step & 1;
        if (step + 1 < NSTEP) stage(1 - p, (step + 1) * 32); // prefetch next tile

        v16bf afrag[2];
        #pragma unroll
        for (int t = 0; t < 2; ++t) {
            int row = wm * 32 + t * 16 + lr;
            v8bf lo = *(const v8bf*)&As[p][row * LDA + kbse];
            v8bf hi = *(const v8bf*)&As[p][row * LDA + 16 + kbse];
            afrag[t] = cat16(lo, hi);
        }
        v16bf bfrag[4];
        #pragma unroll
        for (int u = 0; u < 4; ++u) {
            int nn = wn * 64 + u * 16 + lr;
            v8bf lo = *(const v8bf*)&Bs[p][nn * LDA + kbse];
            v8bf hi = *(const v8bf*)&Bs[p][nn * LDA + 16 + kbse];
            bfrag[u] = cat16(lo, hi);
        }
        #pragma unroll
        for (int t = 0; t < 2; ++t)
            #pragma unroll
            for (int u = 0; u < 4; ++u)
                acc[t][u] = __builtin_amdgcn_wmma_f32_16x16x32_bf16(
                    false, afrag[t], false, bfrag[u], (short)0, acc[t][u], false, false);

        wait_async_all();   // own async fills of buf[1-p] done
        __syncthreads();    // everyone done reading buf[p] & all fills visible
    }

    // Epilogue: z is block-uniform -> one scalar branch, clean store loops.
    if (z == 2) {
        #pragma unroll
        for (int t = 0; t < 2; ++t)
            #pragma unroll
            for (int u = 0; u < 4; ++u)
                #pragma unroll
                for (int r = 0; r < 8; ++r) {
                    int m = m0 + wm * 32 + t * 16 + r + 8 * lh;
                    int n = n0 + wn * 64 + u * 16 + lr;
                    int b = m >> 11, s = m & 2047;
                    vtout[((size_t)b * Ee + n) * Ss + s] = (__bf16)acc[t][u][r];
                }
    } else {
        __bf16* o = (z == 0) ? qout : kout;
        #pragma unroll
        for (int t = 0; t < 2; ++t)
            #pragma unroll
            for (int u = 0; u < 4; ++u)
                #pragma unroll
                for (int r = 0; r < 8; ++r) {
                    int m = m0 + wm * 32 + t * 16 + r + 8 * lh;
                    int n = n0 + wn * 64 + u * 16 + lr;
                    o[(size_t)m * Ee + n] = (__bf16)acc[t][u][r];
                }
    }
}

// ---------- Kernel 4: flash attention (no 1/sqrt(d) scale, as in reference) ----------
// Block = 128 threads (4 waves), 16 query rows per block.
// Wave w: S-tile for 16 keys [kt+16w, kt+16w+16), O-cols [128w, 128w+128).
__global__ __launch_bounds__(128) void attn_kernel(const __bf16* __restrict__ qb,
                                                   const __bf16* __restrict__ kb,
                                                   const __bf16* __restrict__ vt,
                                                   float* __restrict__ out) {
    constexpr int LQ = 520; // padded strides
    constexpr int LP = 72;
    __shared__ __bf16 Qs[16 * LQ];
    __shared__ __bf16 Ps[16 * LP];
    __shared__ float  red[4 * 16];

    const int b  = blockIdx.y;
    const int q0 = blockIdx.x * 16;
    const int tid  = threadIdx.x;
    const int w    = tid >> 5;
    const int lane = tid & 31;
    const int lr   = lane & 15;
    const int lh   = lane >> 4;
    const int kbse = lh * 8;
    const int klo  = lh * 16;

    // Stage Q tile (16 x 512 bf16): 1024 x 16B chunks / 128 threads
    #pragma unroll
    for (int i = 0; i < 8; ++i) {
        int c   = tid + i * 128;
        int row = c >> 6;
        int e8  = c & 63;
        *(uint4*)&Qs[row * LQ + e8 * 8] =
            *(const uint4*)&qb[(size_t)(b * Ss + q0 + row) * Ee + e8 * 8];
    }
    __syncthreads();

    float mrow[8], lrow[8];
    #pragma unroll
    for (int r = 0; r < 8; ++r) { mrow[r] = -3.0e38f; lrow[r] = 0.0f; }
    v8f accO[8] = {};

    for (int kt = 0; kt < Ss; kt += 64) {
        // prefetch next key tile's K rows (lowers to global_prefetch_b8)
        if (kt + 64 < Ss)
            __builtin_prefetch(kb + (size_t)(b * Ss + kt + 64 + w * 16 + lr) * Ee, 0, 3);

        // ---- S = Q . K^T for this wave's 16 keys (fp32 acc over E=512) ----
        v8f s = {};
        const __bf16* krow = kb + (size_t)(b * Ss + kt + w * 16 + lr) * Ee;
        #pragma unroll 4
        for (int e = 0; e < Ee; e += 32) {
            v8bf alo = *(const v8bf*)&Qs[lr * LQ + e + kbse];
            v8bf ahi = *(const v8bf*)&Qs[lr * LQ + e + 16 + kbse];
            v16bf a  = cat16(alo, ahi);
            v8bf blo = *(const v8bf*)&krow[e + kbse];
            v8bf bhi = *(const v8bf*)&krow[e + 16 + kbse];
            v16bf bm = cat16(blo, bhi);
            s = __builtin_amdgcn_wmma_f32_16x16x32_bf16(
                false, a, false, bm, (short)0, s, false, false);
        }

        // ---- per-row tile max (16 lanes of a half hold one row) ----
        float tmax[8];
        #pragma unroll
        for (int r = 0; r < 8; ++r) tmax[r] = rmax16(s[r]);
        if (lr == 0) {
            #pragma unroll
            for (int r = 0; r < 8; ++r) red[w * 16 + r + 8 * lh] = tmax[r];
        }
        __syncthreads(); // (A) red visible; prior-iter Ps/red reads complete via (C)

        // ---- online softmax update ----
        #pragma unroll
        for (int r = 0; r < 8; ++r) {
            int row = r + 8 * lh;
            float t = fmaxf(fmaxf(red[row], red[16 + row]),
                            fmaxf(red[32 + row], red[48 + row]));
            float mnew  = fmaxf(mrow[r], t);
            float alpha = __builtin_amdgcn_exp2f((mrow[r] - mnew) * LOG2E);
            mrow[r] = mnew;
            lrow[r] *= alpha;
            #pragma unroll
            for (int u = 0; u < 8; ++u) accO[u][r] *= alpha;
            s[r] = __builtin_amdgcn_exp2f((s[r] - mnew) * LOG2E); // P
        }
        #pragma unroll
        for (int r = 0; r < 8; ++r) lrow[r] += rsum16(s[r]);

        // ---- P (16x64) -> LDS bf16, A-fragment friendly row-major ----
        #pragma unroll
        for (int r = 0; r < 8; ++r)
            Ps[(r + 8 * lh) * LP + w * 16 + lr] = (__bf16)s[r];
        __syncthreads(); // (B) Ps visible

        // ---- O += P . V  (wave's 128 E-cols; V^T gives contiguous B-frags) ----
        #pragma unroll
        for (int ks = 0; ks < 2; ++ks) {
            v8bf plo = *(const v8bf*)&Ps[lr * LP + ks * 32 + kbse];
            v8bf phi = *(const v8bf*)&Ps[lr * LP + ks * 32 + 16 + kbse];
            v16bf pa = cat16(plo, phi);
            #pragma unroll
            for (int u = 0; u < 8; ++u) {
                int e = w * 128 + u * 16 + lr;
                const __bf16* vrow = vt + ((size_t)b * Ee + e) * Ss + kt + ks * 32 + kbse;
                v8bf vlo = *(const v8bf*)&vrow[0];
                v8bf vhi = *(const v8bf*)&vrow[16];
                v16bf vb = cat16(vlo, vhi);
                accO[u] = __builtin_amdgcn_wmma_f32_16x16x32_bf16(
                    false, pa, false, vb, (short)0, accO[u], false, false);
            }
        }
        __syncthreads(); // (C) protect Ps/red for next iteration
    }

    // ---- normalize and store fp32 output ----
    #pragma unroll
    for (int r = 0; r < 8; ++r) {
        int row = r + 8 * lh;
        float inv = 1.0f / lrow[r];
        #pragma unroll
        for (int u = 0; u < 8; ++u)
            out[(size_t)(b * Ss + q0 + row) * Ee + w * 128 + u * 16 + lr] =
                accO[u][r] * inv;
    }
}

extern "C" void kernel_launch(void* const* d_in, const int* in_sizes, int n_in,
                              void* d_out, int out_size, void* d_ws, size_t ws_size,
                              hipStream_t stream) {
    const float* x  = (const float*)d_in[0];
    const float* wq = (const float*)d_in[1];
    const float* wk = (const float*)d_in[2];
    const float* wv = (const float*)d_in[3];
    float* out = (float*)d_out;

    char* ws = (char*)d_ws;
    __bf16* xb  = (__bf16*)(ws + OFF_XBF);
    __bf16* wt  = (__bf16*)(ws + OFF_WT);
    __bf16* qb  = (__bf16*)(ws + OFF_Q);
    __bf16* kbf = (__bf16*)(ws + OFF_K);
    __bf16* vtb = (__bf16*)(ws + OFF_VT);

    const int n4 = MM * Dd / 4; // 2,097,152
    cvt_x_kernel<<<(n4 + 255) / 256, 256, 0, stream>>>(x, xb, n4);
    cvt_wt_kernel<<<dim3(Ee, 3), 256, 0, stream>>>(wq, wk, wv, wt);
    qkv_gemm_kernel<<<dim3(Ee / 128, MM / 128, 3), 256, 0, stream>>>(xb, wt, qb, kbf, vtb);
    attn_kernel<<<dim3(Ss / 16, Bb), 128, 0, stream>>>(qb, kbf, vtb, out);
}
